// MultiAttrGAT_39917426049525
// MI455X (gfx1250) — matrix operand order repaired and checked by reading, not compiled, not run on hardware
//
#include <hip/hip_runtime.h>
#include <hip/hip_bf16.h>

typedef __attribute__((ext_vector_type(16))) _Float16 v16h;
typedef __attribute__((ext_vector_type(8)))  _Float16 v8h;
typedef __attribute__((ext_vector_type(8)))  float    v8f;
typedef __attribute__((ext_vector_type(4)))  int      v4i;

#if defined(__has_builtin)
#if __has_builtin(__builtin_amdgcn_global_load_async_to_lds_b128)
#define ASYNC_LDS_OK 1
#endif
#endif

// ---------- helpers ----------
__device__ __forceinline__ unsigned int encf(float f) {
    unsigned int u = __float_as_uint(f);
    return (u & 0x80000000u) ? ~u : (u | 0x80000000u);   // order-preserving float->uint
}
__device__ __forceinline__ float decf(unsigned int e) {
    unsigned int u = (e & 0x80000000u) ? (e & 0x7fffffffu) : ~e;
    return __uint_as_float(u);
}
__device__ __forceinline__ float eluf(float v) { return v > 0.f ? v : (__expf(v) - 1.f); }
__device__ __forceinline__ float lrelu(float v) { return v > 0.f ? v : 0.2f * v; }

// ---------- weight prep: fp32 -> fp16 WMMA B-fragments ----------
// Layout: wfrag[mat][ntile][lane][32 halfs]; mat in {Wl1,Wr1,Wl2,Wr2}, K padded to 64.
// Per lane (n = ntile*16 + (lane&15), kb2 = (lane>>4)*16):
//   j<16:  W[(kb2+j)*64 + n]        (K-block 0..31 portion)
//   j>=16: W[(32+kb2+(j-16))*64+n]  (K-block 32..63 portion)
__global__ void prep_weights_k(const float* __restrict__ Wl1, const float* __restrict__ Wr1,
                               const float* __restrict__ Wl2, const float* __restrict__ Wr2,
                               _Float16* __restrict__ wfrag) {
    int tid = blockIdx.x * blockDim.x + threadIdx.x;       // 4*4*32*32 = 16384
    if (tid >= 4 * 4 * 32 * 32) return;
    int mat   = tid >> 12;
    int ntile = (tid >> 10) & 3;
    int lane  = (tid >> 5) & 31;
    int j     = tid & 31;
    const float* src; int K;
    if      (mat == 0) { src = Wl1; K = 40; }
    else if (mat == 1) { src = Wr1; K = 40; }
    else if (mat == 2) { src = Wl2; K = 64; }
    else               { src = Wr2; K = 64; }
    int n   = ntile * 16 + (lane & 15);
    int kb2 = (lane >> 4) * 16;
    int kk  = (j < 16) ? (kb2 + j) : (32 + kb2 + (j - 16));
    float v = (kk < K) ? src[kk * 64 + n] : 0.f;
    wfrag[tid] = (_Float16)v;
}

// ---------- embed + concat -> fp16 [N][64] (cols 40..63 zero) ----------
__global__ void embed_concat_k(const float* __restrict__ xc,
                               const int* __restrict__ hwy, const int* __restrict__ lan,
                               const int* __restrict__ onw,
                               const float* __restrict__ hwyE, const float* __restrict__ lanE,
                               const float* __restrict__ onwE,
                               _Float16* __restrict__ xh, int N) {
    int n = blockIdx.x * blockDim.x + threadIdx.x;
    if (n >= N) return;
    _Float16* o = xh + (size_t)n * 64;
    const float* c = xc + (size_t)n * 12;
    #pragma unroll
    for (int i = 0; i < 12; ++i) o[i] = (_Float16)c[i];
    int h = hwy[n];
    #pragma unroll
    for (int i = 0; i < 16; ++i) o[12 + i] = (_Float16)hwyE[h * 16 + i];
    int l = lan[n];
    #pragma unroll
    for (int i = 0; i < 8; ++i) o[28 + i] = (_Float16)lanE[l * 8 + i];
    int w = onw[n];
    #pragma unroll
    for (int i = 0; i < 4; ++i) o[36 + i] = (_Float16)onwE[w * 4 + i];
    #pragma unroll
    for (int i = 40; i < 64; ++i) o[i] = (_Float16)0.f;
}

// ---------- dual GEMM via WMMA: out{L,R}[N][64] = Xh[N][64] @ W{l,r}[64][64] + b ----------
// block = 256 threads = 8 waves handling one 16-row tile; wave w: mat=w>>2, coltile=w&3.
__global__ void __launch_bounds__(256)
gemm_dual64_k(const _Float16* __restrict__ Xh,     // [N][64] fp16
              const _Float16* __restrict__ Wfrag,  // [2][4][32][32] fp16 fragments (L then R)
              const float* __restrict__ bL, const float* __restrict__ bR,
              float* __restrict__ outL, float* __restrict__ outR, int N) {
    __shared__ __align__(16) _Float16 As[16 * 64];         // 2 KB A tile
    const int row0 = blockIdx.x * 16;
    const char* gsrc = (const char*)(Xh + (size_t)row0 * 64);
#if ASYNC_LDS_OK
    if (threadIdx.x < 128) {   // 128 lanes x 16B = 2 KB, ASYNCcnt-tracked DMA into LDS
        __attribute__((address_space(1))) v4i* g =
            (__attribute__((address_space(1))) v4i*)
                (unsigned long long)(gsrc + threadIdx.x * 16);
        __attribute__((address_space(3))) v4i* l =
            (__attribute__((address_space(3))) v4i*)
                (unsigned int)(unsigned long long)((char*)As + threadIdx.x * 16);
        __builtin_amdgcn_global_load_async_to_lds_b128(g, l, 0, 0);
        asm volatile("s_wait_asynccnt 0x0" ::: "memory");
    }
#else
    ((unsigned long long*)As)[threadIdx.x] = ((const unsigned long long*)gsrc)[threadIdx.x];
#endif
    __syncthreads();

    const int wid  = threadIdx.x >> 5;
    const int lane = threadIdx.x & 31;
    const int mat  = wid >> 2;                 // 0 -> L, 1 -> R
    const int nt   = wid & 3;

    const float* bias = mat ? bR : bL;
    float*       outp = mat ? outR : outL;

    const int m  = lane & 15;
    const int kb = (lane >> 4) * 16;           // A frag K base (0 / 16)
    const int n  = nt * 16 + (lane & 15);

    // A fragments: four 16B LDS loads, combined pairwise
    const _Float16* ar = As + m * 64;
    v8h a0lo = *(const v8h*)(ar + kb);
    v8h a0hi = *(const v8h*)(ar + kb + 8);
    v8h a1lo = *(const v8h*)(ar + 32 + kb);
    v8h a1hi = *(const v8h*)(ar + 32 + kb + 8);
    v16h a0 = __builtin_shufflevector(a0lo, a0hi, 0,1,2,3,4,5,6,7,8,9,10,11,12,13,14,15);
    v16h a1 = __builtin_shufflevector(a1lo, a1hi, 0,1,2,3,4,5,6,7,8,9,10,11,12,13,14,15);

    // B fragments: pre-shuffled, contiguous 64B per lane -> four 16B global loads
    const _Float16* wf = Wfrag + (((size_t)(mat * 4 + nt) * 32 + lane) << 5);
    v8h b0lo = *(const v8h*)(wf);
    v8h b0hi = *(const v8h*)(wf + 8);
    v8h b1lo = *(const v8h*)(wf + 16);
    v8h b1hi = *(const v8h*)(wf + 24);
    v16h b0 = __builtin_shufflevector(b0lo, b0hi, 0,1,2,3,4,5,6,7,8,9,10,11,12,13,14,15);
    v16h b1 = __builtin_shufflevector(b1lo, b1hi, 0,1,2,3,4,5,6,7,8,9,10,11,12,13,14,15);

    v8f c = {};
    c = __builtin_amdgcn_wmma_f32_16x16x32_f16(false, a0, false, b0, (short)0, c, false, false);
    c = __builtin_amdgcn_wmma_f32_16x16x32_f16(false, a1, false, b1, (short)0, c, false, false);

    const float bn    = bias[n];
    const int   mbase = (lane >> 4) * 8;
    float* basep = outp + (size_t)(row0 + mbase) * 64 + n;
    if (row0 + 16 <= N) {                       // uniform fast path: immediate-offset stores
        #pragma unroll
        for (int j = 0; j < 8; ++j) basep[j * 64] = c[j] + bn;
    } else {
        #pragma unroll
        for (int j = 0; j < 8; ++j)
            if (row0 + mbase + j < N) basep[j * 64] = c[j] + bn;
    }
}

// ---------- edge pass 1: GATv2 logits + segment max (ordered-uint atomicMax) ----------
__global__ void edge_logit_max_k(const int* __restrict__ esrc, const int* __restrict__ edst,
                                 int E, int Etot,
                                 const float* __restrict__ xl, const float* __restrict__ xr,
                                 const float* __restrict__ att,        // [2][32]
                                 float* __restrict__ logitb,           // [Etot][2]
                                 unsigned int* __restrict__ mEnc) {    // [N][2]
    int e = blockIdx.x * blockDim.x + threadIdx.x;
    if (e >= Etot) return;
    int s, d;
    if (e < E) { s = esrc[e]; d = edst[e]; } else { s = d = e - E; }
    const float4* xs = (const float4*)(xl + (size_t)s * 64);
    const float4* xd = (const float4*)(xr + (size_t)d * 64);
    const float4* aw = (const float4*)att;
    float lg[2] = {0.f, 0.f};
    #pragma unroll
    for (int i = 0; i < 16; ++i) {
        float4 a = xs[i], b = xd[i], w = aw[i];
        int h = i >> 3;
        lg[h] += w.x * lrelu(a.x + b.x);
        lg[h] += w.y * lrelu(a.y + b.y);
        lg[h] += w.z * lrelu(a.z + b.z);
        lg[h] += w.w * lrelu(a.w + b.w);
    }
    logitb[(size_t)e * 2 + 0] = lg[0];
    logitb[(size_t)e * 2 + 1] = lg[1];
    atomicMax(&mEnc[(size_t)d * 2 + 0], encf(lg[0]));
    atomicMax(&mEnc[(size_t)d * 2 + 1], encf(lg[1]));
}

// ---------- edge pass 2: a = exp(logit - max); segment sum ----------
__global__ void edge_exp_sum_k(const int* __restrict__ edst, int E, int Etot,
                               float* __restrict__ logitb,
                               const unsigned int* __restrict__ mEnc,
                               float* __restrict__ denom) {
    int e = blockIdx.x * blockDim.x + threadIdx.x;
    if (e >= Etot) return;
    int d = (e < E) ? edst[e] : (e - E);
    #pragma unroll
    for (int h = 0; h < 2; ++h) {
        float lg = logitb[(size_t)e * 2 + h];
        float a  = __expf(lg - decf(mEnc[(size_t)d * 2 + h]));
        logitb[(size_t)e * 2 + h] = a;
        unsafeAtomicAdd(&denom[(size_t)d * 2 + h], a);
    }
}

// ---------- edge pass 3: accum[dst] += alpha * xl[src] ----------
__global__ void edge_scatter_k(const int* __restrict__ esrc, const int* __restrict__ edst,
                               int E, int Etot,
                               const float* __restrict__ xl,
                               const float* __restrict__ logitb,
                               const float* __restrict__ denom,
                               float* __restrict__ accum) {
    int e = blockIdx.x * blockDim.x + threadIdx.x;
    if (e >= Etot) return;
    int s, d;
    if (e < E) { s = esrc[e]; d = edst[e]; } else { s = d = e - E; }
    __builtin_prefetch(xl + (size_t)s * 64, 0, 1);          // global_prefetch_b8
    float al[2];
    #pragma unroll
    for (int h = 0; h < 2; ++h)
        al[h] = logitb[(size_t)e * 2 + h] / (denom[(size_t)d * 2 + h] + 1e-16f);
    const float4* xs = (const float4*)(xl + (size_t)s * 64);
    float* ap = accum + (size_t)d * 64;
    #pragma unroll
    for (int i = 0; i < 16; ++i) {
        float4 v = xs[i];
        float  a = al[i >> 3];
        unsafeAtomicAdd(ap + i * 4 + 0, a * v.x);
        unsafeAtomicAdd(ap + i * 4 + 1, a * v.y);
        unsafeAtomicAdd(ap + i * 4 + 2, a * v.z);
        unsafeAtomicAdd(ap + i * 4 + 3, a * v.w);
    }
}

// ---------- layer-1 epilogue: ELU(accum + bias1) -> fp16 features for layer 2 ----------
__global__ void post_layer1_k(const float* __restrict__ accum, const float* __restrict__ bias1,
                              _Float16* __restrict__ xh, int N) {
    int t = blockIdx.x * blockDim.x + threadIdx.x;
    if (t >= N * 64) return;
    int c = t & 63;
    xh[t] = (_Float16)eluf(accum[t] + bias1[c]);
}

// ---------- layer-2 epilogue: ELU(mean-over-heads + bias2) -> h2 [N][32] ----------
__global__ void post_layer2_k(const float* __restrict__ accum, const float* __restrict__ bias2,
                              float* __restrict__ h2, int N) {
    int t = blockIdx.x * blockDim.x + threadIdx.x;
    if (t >= N * 32) return;
    int n = t >> 5, c = t & 31;
    float v = 0.5f * (accum[(size_t)n * 64 + c] + accum[(size_t)n * 64 + 32 + c]) + bias2[c];
    h2[t] = eluf(v);
}

// ---------- prediction heads ----------
__global__ void heads_k(const float* __restrict__ h2,
                        const float* __restrict__ Whwy, const float* __restrict__ bhwy,
                        const float* __restrict__ Wlan, const float* __restrict__ blan,
                        const float* __restrict__ Wonw, const float* __restrict__ bonw,
                        const float* __restrict__ Wwid, const float* __restrict__ bwid,
                        const float* __restrict__ Wmax, const float* __restrict__ bmax,
                        const float* __restrict__ Wmin, const float* __restrict__ bmin,
                        float* __restrict__ out, int N) {
    int n = blockIdx.x * blockDim.x + threadIdx.x;
    if (n >= N) return;
    float h[32];
    const float4* hp = (const float4*)(h2 + (size_t)n * 32);
    #pragma unroll
    for (int i = 0; i < 8; ++i) {
        float4 v = hp[i];
        h[i * 4 + 0] = v.x; h[i * 4 + 1] = v.y; h[i * 4 + 2] = v.z; h[i * 4 + 3] = v.w;
    }
    #pragma unroll
    for (int j = 0; j < 16; ++j) {
        float a = bhwy[j];
        #pragma unroll
        for (int c = 0; c < 32; ++c) a += h[c] * Whwy[c * 16 + j];
        out[(size_t)n * 16 + j] = a;
    }
    float* o1 = out + (size_t)16 * N;
    #pragma unroll
    for (int j = 0; j < 3; ++j) {
        float a = blan[j];
        #pragma unroll
        for (int c = 0; c < 32; ++c) a += h[c] * Wlan[c * 3 + j];
        o1[(size_t)n * 3 + j] = a;
    }
    float a2 = bonw[0], a3 = bwid[0], a4 = bmax[0], a5 = bmin[0];
    #pragma unroll
    for (int c = 0; c < 32; ++c) {
        float hv = h[c];
        a2 += hv * Wonw[c]; a3 += hv * Wwid[c]; a4 += hv * Wmax[c]; a5 += hv * Wmin[c];
    }
    out[(size_t)19 * N + n] = a2;
    out[(size_t)20 * N + n] = a3;
    out[(size_t)21 * N + n] = a4;
    out[(size_t)22 * N + n] = a5;
}

// ---------- launch ----------
extern "C" void kernel_launch(void* const* d_in, const int* in_sizes, int n_in,
                              void* d_out, int out_size, void* d_ws, size_t ws_size,
                              hipStream_t stream) {
    const float* x_cont  = (const float*)d_in[0];
    const int*   hwy_in  = (const int*)d_in[1];
    const int*   lan_in  = (const int*)d_in[2];
    const int*   onw_in  = (const int*)d_in[3];
    const int*   eidx    = (const int*)d_in[4];
    const float* hwyE    = (const float*)d_in[5];
    const float* lanE    = (const float*)d_in[6];
    const float* onwE    = (const float*)d_in[7];
    const float* Wl1 = (const float*)d_in[8];  const float* bl1 = (const float*)d_in[9];
    const float* Wr1 = (const float*)d_in[10]; const float* br1 = (const float*)d_in[11];
    const float* att1 = (const float*)d_in[12]; const float* bias1 = (const float*)d_in[13];
    const float* Wl2 = (const float*)d_in[14]; const float* bl2 = (const float*)d_in[15];
    const float* Wr2 = (const float*)d_in[16]; const float* br2 = (const float*)d_in[17];
    const float* att2 = (const float*)d_in[18]; const float* bias2 = (const float*)d_in[19];
    const float* Whwy = (const float*)d_in[20]; const float* bhwy = (const float*)d_in[21];
    const float* Wlan = (const float*)d_in[22]; const float* blan = (const float*)d_in[23];
    const float* Wonw = (const float*)d_in[24]; const float* bonw = (const float*)d_in[25];
    const float* Wwid = (const float*)d_in[26]; const float* bwid = (const float*)d_in[27];
    const float* Wmax = (const float*)d_in[28]; const float* bmax = (const float*)d_in[29];
    const float* Wmin = (const float*)d_in[30]; const float* bmin = (const float*)d_in[31];

    const int N    = in_sizes[1];
    const int E    = in_sizes[4] / 2;
    const int Etot = E + N;
    const int* esrc = eidx;
    const int* edst = eidx + E;

    // workspace carve (256B aligned)
    char* p = (char*)d_ws;
    auto carve = [&](size_t bytes) {
        void* r = (void*)p;
        p += (bytes + 255) & ~(size_t)255;
        return r;
    };
    _Float16*     xh     = (_Float16*)carve((size_t)N * 64 * sizeof(_Float16));
    _Float16*     wfrag  = (_Float16*)carve((size_t)4 * 4 * 32 * 32 * sizeof(_Float16));
    float*        xl     = (float*)carve((size_t)N * 64 * sizeof(float));
    float*        xr     = (float*)carve((size_t)N * 64 * sizeof(float));
    float*        logitb = (float*)carve((size_t)Etot * 2 * sizeof(float));
    unsigned int* mEnc   = (unsigned int*)carve((size_t)N * 2 * sizeof(unsigned int));
    float*        denom  = (float*)carve((size_t)N * 2 * sizeof(float));
    float*        accum  = (float*)carve((size_t)N * 64 * sizeof(float));
    float*        h2     = (float*)carve((size_t)N * 32 * sizeof(float));

    const int nb  = (N + 255) / 256;
    const int nt  = (N + 15) / 16;
    const int eb  = (Etot + 255) / 256;
    const int n64 = ((N * 64) + 255) / 256;
    const int n32 = ((N * 32) + 255) / 256;

    prep_weights_k<<<(4 * 4 * 32 * 32 + 255) / 256, 256, 0, stream>>>(Wl1, Wr1, Wl2, Wr2, wfrag);
    embed_concat_k<<<nb, 256, 0, stream>>>(x_cont, hwy_in, lan_in, onw_in, hwyE, lanE, onwE, xh, N);

    // ===== layer 1 =====
    gemm_dual64_k<<<nt, 256, 0, stream>>>(xh, wfrag, bl1, br1, xl, xr, N);
    (void)hipMemsetAsync(mEnc, 0, (size_t)N * 2 * sizeof(unsigned int), stream);
    (void)hipMemsetAsync(denom, 0, (size_t)N * 2 * sizeof(float), stream);
    (void)hipMemsetAsync(accum, 0, (size_t)N * 64 * sizeof(float), stream);
    edge_logit_max_k<<<eb, 256, 0, stream>>>(esrc, edst, E, Etot, xl, xr, att1, logitb, mEnc);
    edge_exp_sum_k<<<eb, 256, 0, stream>>>(edst, E, Etot, logitb, mEnc, denom);
    edge_scatter_k<<<eb, 256, 0, stream>>>(esrc, edst, E, Etot, xl, logitb, denom, accum);
    post_layer1_k<<<n64, 256, 0, stream>>>(accum, bias1, xh, N);

    // ===== layer 2 =====
    gemm_dual64_k<<<nt, 256, 0, stream>>>(xh, wfrag + (size_t)2 * 4 * 32 * 32, bl2, br2, xl, xr, N);
    (void)hipMemsetAsync(mEnc, 0, (size_t)N * 2 * sizeof(unsigned int), stream);
    (void)hipMemsetAsync(denom, 0, (size_t)N * 2 * sizeof(float), stream);
    (void)hipMemsetAsync(accum, 0, (size_t)N * 64 * sizeof(float), stream);
    edge_logit_max_k<<<eb, 256, 0, stream>>>(esrc, edst, E, Etot, xl, xr, att2, logitb, mEnc);
    edge_exp_sum_k<<<eb, 256, 0, stream>>>(edst, E, Etot, logitb, mEnc, denom);
    edge_scatter_k<<<eb, 256, 0, stream>>>(esrc, edst, E, Etot, xl, logitb, denom, accum);
    post_layer2_k<<<n32, 256, 0, stream>>>(accum, bias2, h2, N);

    // ===== heads =====
    heads_k<<<nb, 256, 0, stream>>>(h2, Whwy, bhwy, Wlan, blan, Wonw, bonw,
                                    Wwid, bwid, Wmax, bmax, Wmin, bmin,
                                    (float*)d_out, N);
}
